// GPT_74620761801194
// MI455X (gfx1250) — matrix operand order repaired and checked by reading, not compiled, not run on hardware
//
#include <hip/hip_runtime.h>
#include <math.h>

// ---------------------------------------------------------------------------
// Model constants (from the reference)
// ---------------------------------------------------------------------------
#define C_EMB   768
#define NH      12
#define HD      64
#define T_HALF  512
#define BATCH   4
#define M_TOK   (BATCH * T_HALF)     // 2048 token rows per side
#define VOCAB_N 50304
#define BLOCK_FULL 1024

// ---------------------------------------------------------------------------
// WMMA types (gfx1250, wave32)
// ---------------------------------------------------------------------------
typedef _Float16 f16;
typedef __attribute__((ext_vector_type(16))) _Float16 v16h;
typedef __attribute__((ext_vector_type(8)))  _Float16 v8h;
typedef __attribute__((ext_vector_type(8)))  float    v8f;

union U16 { v8h h[2]; v16h v; };

// LDS tile row stride in halfs (32 data + 8 pad -> 80B rows, keeps 16B align)
#define LDA_T 40

static __device__ __forceinline__ v8f wmma_f16(const U16& a, const U16& b, v8f c) {
  return __builtin_amdgcn_wmma_f32_16x16x32_f16(false, a.v, false, b.v,
                                                (short)0, c, false, false);
}

// ---------------------------------------------------------------------------
// GEMM: C[M x N] (+=) A[M x K] * W[N x K]^T + bias[N]
// A, W fp32 row-major (lda = ldw = K); C fp32 with row stride ldc.
// Requires K % 32 == 0 and N % 128 == 0 (true for every call); M guarded when
// !FULL (branchless clamp+mask). Block tile 128x128x32, ping-pong LDS, 8
// waves, each wave a 32x64 tile = 8 x v_wmma_f32_16x16x32_f16 per K-step.
// ---------------------------------------------------------------------------
template <bool FULL, bool ACC>
__global__ __launch_bounds__(256) void gemm_wmma(
    const float* __restrict__ A, const float* __restrict__ W,
    const float* __restrict__ bias, float* __restrict__ Co,
    int M, int N, int K, int ldc)
{
  __shared__ __align__(16) f16 As[2][128 * LDA_T];
  __shared__ __align__(16) f16 Bs[2][128 * LDA_T];

  const int tid = threadIdx.x;
  const int m0  = blockIdx.y * 128;
  const int n0  = blockIdx.x * 128;

  // cooperative-staging mapping: 2 threads/row, 16 floats each
  const int srow = tid >> 1;
  const int scol = (tid & 1) << 4;

  auto stage = [&](int buf, int k0) {
    { // A tile (128 x 32)
      const int gr = m0 + srow;
      int grc = gr; float msk = 1.f;
      if (!FULL) { grc = (gr < M) ? gr : (M - 1); msk = (gr < M) ? 1.f : 0.f; }
      const float* ap = A + (size_t)grc * K + k0 + scol;
      f16* dst = &As[buf][srow * LDA_T + scol];
      #pragma unroll
      for (int u = 0; u < 16; u += 4) {
        float4 f = *(const float4*)(ap + u);
        if (!FULL) { f.x *= msk; f.y *= msk; f.z *= msk; f.w *= msk; }
        dst[u + 0] = (f16)f.x; dst[u + 1] = (f16)f.y;
        dst[u + 2] = (f16)f.z; dst[u + 3] = (f16)f.w;
      }
    }
    { // B tile (128 x 32): rows of W (N index); N % 128 == 0 always
      const int gn = n0 + srow;
      const float* wp = W + (size_t)gn * K + k0 + scol;
      f16* dst = &Bs[buf][srow * LDA_T + scol];
      #pragma unroll
      for (int u = 0; u < 16; u += 4) {
        float4 f = *(const float4*)(wp + u);
        dst[u + 0] = (f16)f.x; dst[u + 1] = (f16)f.y;
        dst[u + 2] = (f16)f.z; dst[u + 3] = (f16)f.w;
      }
    }
  };

  // wave / fragment mapping (per ISA 16-bit A 16x32 / B 32x16 layouts)
  const int lane = tid & 31;
  const int wv   = tid >> 5;
  const int wm   = wv & 3;            // 4 waves along M (32 rows each)
  const int wn   = wv >> 2;           // 2 waves along N (64 cols each)
  const int mrow = lane & 15;
  const int kA   = (lane >> 4) << 3;  // 0 or 8
  const int nIdx = lane & 15;
  const int kB   = (lane >> 4) << 3;  // 0 or 8

  v8f acc[2][4] = {};

  stage(0, 0);
  int cur = 0;
  for (int k0 = 0; k0 < K; k0 += 32) {
    __syncthreads();
    if (k0 + 32 < K) stage(cur ^ 1, k0 + 32);

    U16 a[2], b[4];
    #pragma unroll
    for (int mi = 0; mi < 2; ++mi) {
      const f16* p = &As[cur][(wm * 32 + mi * 16 + mrow) * LDA_T];
      a[mi].h[0] = *(const v8h*)(p + kA);
      a[mi].h[1] = *(const v8h*)(p + 16 + kA);
    }
    #pragma unroll
    for (int ni = 0; ni < 4; ++ni) {
      const f16* p = &Bs[cur][(wn * 64 + ni * 16 + nIdx) * LDA_T + kB];
      b[ni].h[0] = *(const v8h*)(p);
      b[ni].h[1] = *(const v8h*)(p + 16);
    }
    #pragma unroll
    for (int mi = 0; mi < 2; ++mi)
      #pragma unroll
      for (int ni = 0; ni < 4; ++ni)
        acc[mi][ni] = wmma_f16(a[mi], b[ni], acc[mi][ni]);
    cur ^= 1;
  }

  // store: C/D VGPR j -> (M = j + 8*(lane>=16), N = lane&15)
  const int rb0 = m0 + wm * 32 + ((lane >> 4) << 3);
  const int cb0 = n0 + wn * 64 + (lane & 15);
  #pragma unroll
  for (int ni = 0; ni < 4; ++ni) {
    const int cc = cb0 + ni * 16;
    const float bv = bias ? bias[cc] : 0.f;
    #pragma unroll
    for (int mi = 0; mi < 2; ++mi) {
      const int r0 = rb0 + mi * 16;
      #pragma unroll
      for (int j = 0; j < 8; ++j) {
        const int r = r0 + j;
        if (FULL || r < M) {
          size_t o = (size_t)r * ldc + cc;
          float val = acc[mi][ni][j] + bv;
          if (ACC) Co[o] += val; else Co[o] = val;
        }
      }
    }
  }
}

// ---------------------------------------------------------------------------
// WMMA flash self-attention. One workgroup per (b, h, 128-query chunk);
// 8 waves x 16 queries. K/V tiles (32 keys) staged cooperatively in LDS
// (V transposed so PV B-fragments are contiguous). Scores: 2x2 chained
// WMMAs over hd=64; online softmax with 16-lane-half shfl reductions;
// P reshaped D-layout -> A-layout through a private per-wave LDS tile.
// ---------------------------------------------------------------------------
__global__ __launch_bounds__(256) void attn_self_wmma(
    const float* __restrict__ qkv, float* __restrict__ y, int causal)
{
  __shared__ __align__(16) f16 Kl[32 * 80];      // [key][hd], stride 80
  __shared__ __align__(16) f16 Vt[64 * LDA_T];   // [hd][key], stride 40
  __shared__ __align__(16) f16 Pl[8][16 * LDA_T];// per-wave P (16 x 32)

  const int tid  = threadIdx.x;
  const int wv   = tid >> 5;
  const int lane = tid & 31;

  const int tblk = blockIdx.x & 3;
  const int h    = (blockIdx.x >> 2) % NH;
  const int b    = blockIdx.x / (4 * NH);
  const int t0   = tblk * 128 + wv * 16;

  const int mq  = lane & 15;
  const int kAb = (lane >> 4) << 3;   // A-frag K base: 0/8
  const int kBb = (lane >> 4) << 3;   // B-frag K base: 0/8
  const int hi8 = (lane >> 4) << 3;   // row offset owned by this lane half

  // K/V staging map: 8 threads/row, 8 floats each
  const int sKrow = tid >> 3;         // 0..31 (key within tile)
  const int sKcol = (tid & 7) << 3;   // 0..56 (head dim)

  // ---- Q fragments (A operand), pre-scaled by 1/sqrt(hd) ----
  U16 qf[2];
  {
    const float* qp = qkv + (size_t)(b * T_HALF + t0 + mq) * (3 * C_EMB) + h * HD;
    #pragma unroll
    for (int f = 0; f < 2; ++f) {
      #pragma unroll
      for (int i = 0; i < 16; ++i) {
        int k = f * 32 + ((i < 8) ? (kAb + i) : (16 + kAb + (i - 8)));
        qf[f].v[i] = (f16)(qp[k] * 0.125f);
      }
    }
  }

  v8f Y[4] = {};
  float mrow[8], srow[8];
  #pragma unroll
  for (int j = 0; j < 8; ++j) { mrow[j] = -1e30f; srow[j] = 0.f; }

  const int sendBlock = causal ? (tblk * 128 + 128) : T_HALF;  // block-uniform
  const int sendWave  = causal ? (t0 + 16)          : T_HALF;  // wave-uniform

  for (int jb = 0; jb < sendBlock; jb += 32) {
    __syncthreads();    // previous tile fully consumed
    { // ---- stage K tile [key][hd] and V tile transposed [hd][key] ----
      int s = jb + sKrow; if (s >= T_HALF) s = T_HALF - 1;
      const float* kp = qkv + (size_t)(b * T_HALF + s) * (3 * C_EMB) + C_EMB + h * HD + sKcol;
      f16* kd = &Kl[sKrow * 80 + sKcol];
      float4 f0 = *(const float4*)(kp + 0);
      float4 f1 = *(const float4*)(kp + 4);
      kd[0] = (f16)f0.x; kd[1] = (f16)f0.y; kd[2] = (f16)f0.z; kd[3] = (f16)f0.w;
      kd[4] = (f16)f1.x; kd[5] = (f16)f1.y; kd[6] = (f16)f1.z; kd[7] = (f16)f1.w;
      const float* vp = kp + C_EMB;
      float4 g0 = *(const float4*)(vp + 0);
      float4 g1 = *(const float4*)(vp + 4);
      Vt[(sKcol + 0) * LDA_T + sKrow] = (f16)g0.x;
      Vt[(sKcol + 1) * LDA_T + sKrow] = (f16)g0.y;
      Vt[(sKcol + 2) * LDA_T + sKrow] = (f16)g0.z;
      Vt[(sKcol + 3) * LDA_T + sKrow] = (f16)g0.w;
      Vt[(sKcol + 4) * LDA_T + sKrow] = (f16)g1.x;
      Vt[(sKcol + 5) * LDA_T + sKrow] = (f16)g1.y;
      Vt[(sKcol + 6) * LDA_T + sKrow] = (f16)g1.z;
      Vt[(sKcol + 7) * LDA_T + sKrow] = (f16)g1.w;
    }
    __syncthreads();    // staging visible

    if (jb < sendWave) {            // wave-uniform: EXEC stays all-ones
      // ---- scores: S0 = keys jb..+15, S1 = keys jb+16..+31 ----
      v8f S0 = {}, S1 = {};
      U16 kf;
      #pragma unroll
      for (int sub = 0; sub < 2; ++sub) {
        const f16* kr = &Kl[(sub * 16 + (lane & 15)) * 80];
        #pragma unroll
        for (int f = 0; f < 2; ++f) {
          kf.h[0] = *(const v8h*)(kr + f * 32 + kBb);
          kf.h[1] = *(const v8h*)(kr + f * 32 + kBb + 16);
          if (sub == 0) S0 = wmma_f16(qf[f], kf, S0);
          else          S1 = wmma_f16(qf[f], kf, S1);
        }
      }

      // ---- online softmax (row stats live in 16-lane halves) ----
      #pragma unroll
      for (int j = 0; j < 8; ++j) {
        const int r = t0 + j + hi8;
        float s0 = S0[j], s1 = S1[j];
        if (causal) {
          if (jb + (lane & 15) > r)      s0 = -1e30f;
          if (jb + 16 + (lane & 15) > r) s1 = -1e30f;
        }
        float mt = fmaxf(s0, s1);
        #pragma unroll
        for (int o = 1; o < 16; o <<= 1) mt = fmaxf(mt, __shfl_xor(mt, o, 32));
        const float nm = fmaxf(mrow[j], mt);
        const float p0 = __expf(s0 - nm);
        const float p1 = __expf(s1 - nm);
        float rs = p0 + p1;
        #pragma unroll
        for (int o = 1; o < 16; o <<= 1) rs += __shfl_xor(rs, o, 32);
        const float resc = __expf(mrow[j] - nm);
        srow[j] = srow[j] * resc + rs;
        mrow[j] = nm;
        #pragma unroll
        for (int ni = 0; ni < 4; ++ni) Y[ni][j] *= resc;
        f16* pr = &Pl[wv][(j + hi8) * LDA_T];
        pr[lane & 15]        = (f16)p0;
        pr[16 + (lane & 15)] = (f16)p1;
      }

      // ---- PV: P (16x32) x V (32x64) ----
      U16 pf;
      {
        const f16* pp = &Pl[wv][mq * LDA_T];
        pf.h[0] = *(const v8h*)(pp + kAb);
        pf.h[1] = *(const v8h*)(pp + 16 + kAb);
      }
      #pragma unroll
      for (int ni = 0; ni < 4; ++ni) {
        U16 vf;
        const f16* vr = &Vt[(ni * 16 + (lane & 15)) * LDA_T + kBb];
        vf.h[0] = *(const v8h*)(vr);
        vf.h[1] = *(const v8h*)(vr + 16);
        Y[ni] = wmma_f16(pf, vf, Y[ni]);
      }
    }
  }

  // ---- normalize and store ----
  #pragma unroll
  for (int j = 0; j < 8; ++j) {
    const int r = t0 + j + hi8;
    const float inv = 1.f / srow[j];
    float* yp = y + (size_t)(b * T_HALF + r) * C_EMB + h * HD + (lane & 15);
    #pragma unroll
    for (int ni = 0; ni < 4; ++ni) yp[ni * 16] = Y[ni][j] * inv;
  }
}

// ---------------------------------------------------------------------------
// Embedding: x = wte[idx] + wpe[pos]; encoder overwrites last 10 rows w/ mask
// ---------------------------------------------------------------------------
__global__ void embed_kernel(const int* __restrict__ idx,
                             const float* __restrict__ wte,
                             const float* __restrict__ wpe,
                             const float* __restrict__ mask_emb,
                             float* __restrict__ x, int dec)
{
  int i = blockIdx.x * blockDim.x + threadIdx.x;
  if (i >= M_TOK * C_EMB) return;
  int row = i / C_EMB, c = i % C_EMB;
  int b = row / T_HALF, t = row % T_HALF;
  float v;
  if (!dec && t >= T_HALF - 10) {
    v = mask_emb[c];
  } else {
    int tok = idx[b * BLOCK_FULL + (dec ? T_HALF : 0) + t];
    v = wte[(size_t)tok * C_EMB + c] + wpe[(size_t)t * C_EMB + c];
  }
  x[i] = v;
}

// ---------------------------------------------------------------------------
// LayerNorm (row = 768), one block per row
// ---------------------------------------------------------------------------
__global__ __launch_bounds__(256) void ln_kernel(
    const float* __restrict__ x, const float* __restrict__ w,
    const float* __restrict__ bb, float* __restrict__ o)
{
  const int row = blockIdx.x, tid = threadIdx.x;
  const float* xr = x + (size_t)row * C_EMB;
  __shared__ float red[256];

  float s = 0.f;
  for (int c = tid; c < C_EMB; c += 256) s += xr[c];
  red[tid] = s; __syncthreads();
  for (int st = 128; st; st >>= 1) { if (tid < st) red[tid] += red[tid + st]; __syncthreads(); }
  float mean = red[0] / (float)C_EMB; __syncthreads();

  float v = 0.f;
  for (int c = tid; c < C_EMB; c += 256) { float d = xr[c] - mean; v += d * d; }
  red[tid] = v; __syncthreads();
  for (int st = 128; st; st >>= 1) { if (tid < st) red[tid] += red[tid + st]; __syncthreads(); }
  float rstd = rsqrtf(red[0] / (float)C_EMB + 1e-5f);

  for (int c = tid; c < C_EMB; c += 256)
    o[(size_t)row * C_EMB + c] = (xr[c] - mean) * rstd * w[c] + bb[c];
}

__global__ void gelu_kernel(float* __restrict__ h, int n)
{
  int i = blockIdx.x * blockDim.x + threadIdx.x;
  if (i < n) { float x = h[i]; h[i] = 0.5f * x * (1.f + erff(x * 0.70710678118654752f)); }
}

// ---------------------------------------------------------------------------
// Cross-attention: S = 5 memory slots; also writes att[b,h,t,s] for the KL.
// One wave per (b,h,t); trivially small, stays on VALU.
// ---------------------------------------------------------------------------
__global__ __launch_bounds__(256) void attn_cross(
    const float* __restrict__ q, const float* __restrict__ kv,
    float* __restrict__ y, float* __restrict__ att)
{
  const int gw   = blockIdx.x * 8 + (threadIdx.x >> 5);
  const int lane = threadIdx.x & 31;
  const int t = gw % T_HALF;
  const int h = (gw / T_HALF) % NH;
  const int b = gw / (T_HALF * NH);

  const float* qp = q + (size_t)(b * T_HALF + t) * C_EMB + h * HD;
  const float q0 = qp[lane], q1 = qp[lane + 32];

  float sc[5]; float mx = -3.4e38f;
  #pragma unroll
  for (int s = 0; s < 5; ++s) {
    const float* kp = kv + (size_t)(b * 5 + s) * (2 * C_EMB) + h * HD;
    float d = q0 * kp[lane] + q1 * kp[lane + 32];
    #pragma unroll
    for (int o = 16; o; o >>= 1) d += __shfl_xor(d, o, 32);
    d *= 0.125f;
    sc[s] = d; mx = fmaxf(mx, d);
  }
  float sum = 0.f, a0 = 0.f, a1 = 0.f, p[5];
  #pragma unroll
  for (int s = 0; s < 5; ++s) {
    p[s] = __expf(sc[s] - mx); sum += p[s];
    const float* vp = kv + (size_t)(b * 5 + s) * (2 * C_EMB) + C_EMB + h * HD;
    a0 += p[s] * vp[lane];
    a1 += p[s] * vp[lane + 32];
  }
  float* yp = y + (size_t)(b * T_HALF + t) * C_EMB + h * HD;
  yp[lane] = a0 / sum; yp[lane + 32] = a1 / sum;
  if (lane == 0) {
    float* ap = att + ((size_t)(b * NH + h) * T_HALF + t) * 5;
    #pragma unroll
    for (int s = 0; s < 5; ++s) ap[s] = p[s] / sum;
  }
}

// ---------------------------------------------------------------------------
// KL: m = mean_h att -> argmax over S -> per-batch histogram -> KL vs U(5)
// ---------------------------------------------------------------------------
__global__ __launch_bounds__(256) void kl_kernel(
    const float* __restrict__ att, float* __restrict__ out)
{
  __shared__ float hist[BATCH * 5];
  const int tid = threadIdx.x;
  if (tid < BATCH * 5) hist[tid] = 0.f;
  __syncthreads();

  for (int i = tid; i < BATCH * T_HALF; i += 256) {
    int b = i / T_HALF, t = i % T_HALF;
    float m[5] = {0.f, 0.f, 0.f, 0.f, 0.f};
    for (int h = 0; h < NH; ++h) {
      const float* ap = att + ((size_t)(b * NH + h) * T_HALF + t) * 5;
      #pragma unroll
      for (int s = 0; s < 5; ++s) m[s] += ap[s];
    }
    int am = 0; float best = m[0];
    #pragma unroll
    for (int s = 1; s < 5; ++s) if (m[s] > best) { best = m[s]; am = s; }
    atomicAdd(&hist[b * 5 + am], 1.f);
  }
  __syncthreads();

  if (tid == 0) {
    const float log_u = -logf(5.f);
    float kl = 0.f;
    for (int b = 0; b < BATCH; ++b) {
      float tot = 0.f;
      for (int s = 0; s < 5; ++s) tot += hist[b * 5 + s];
      float klb = 0.f;
      for (int s = 0; s < 5; ++s) {
        float hh = hist[b * 5 + s] / tot;
        if (hh > 0.f) klb += hh * (logf(hh) - log_u);
      }
      kl += klb;
    }
    *out = kl / (float)BATCH;
  }
}

__global__ void mean_kernel(const float* __restrict__ x, float* __restrict__ o)
{
  int i = blockIdx.x * blockDim.x + threadIdx.x;
  if (i >= BATCH * C_EMB) return;
  int b = i / C_EMB, c = i % C_EMB;
  float s = 0.f;
  for (int t = 0; t < T_HALF; ++t)
    s += x[(size_t)(b * T_HALF + t) * C_EMB + c];
  o[i] = s / (float)T_HALF;
}

// ---------------------------------------------------------------------------
// Cross-entropy: block per row; log-softmax over 50304; atomic accumulate
// ---------------------------------------------------------------------------
__global__ __launch_bounds__(256) void loss_rows(
    const float* __restrict__ logits, const int* __restrict__ targets,
    float* __restrict__ ce)
{
  const int row = blockIdx.x, tid = threadIdx.x;
  const float* lr = logits + (size_t)row * VOCAB_N;
  __shared__ float red[256];

  float mx = -3.4e38f;
  for (int c = tid; c < VOCAB_N; c += 256) mx = fmaxf(mx, lr[c]);
  red[tid] = mx; __syncthreads();
  for (int st = 128; st; st >>= 1) { if (tid < st) red[tid] = fmaxf(red[tid], red[tid + st]); __syncthreads(); }
  mx = red[0]; __syncthreads();

  float s = 0.f;
  for (int c = tid; c < VOCAB_N; c += 256) s += __expf(lr[c] - mx);
  red[tid] = s; __syncthreads();
  for (int st = 128; st; st >>= 1) { if (tid < st) red[tid] += red[tid + st]; __syncthreads(); }

  if (tid == 0) {
    int b = row / T_HALF, t = row % T_HALF;
    int tgt = targets[b * BLOCK_FULL + T_HALF + t];
    float lp = lr[tgt] - mx - logf(red[0]);
    atomicAdd(ce, -lp);
  }
}

__global__ void zero_kernel(float* p, int n)
{
  int i = blockIdx.x * blockDim.x + threadIdx.x;
  if (i < n) p[i] = 0.f;
}

__global__ void finalize_kernel(const float* __restrict__ ce,
                                const float* __restrict__ kls,
                                float* __restrict__ out)
{
  if (threadIdx.x == 0)
    out[0] = ce[0] / (float)M_TOK + (kls[0] + kls[1] + kls[2]) / 3.f;
}

// ---------------------------------------------------------------------------
// Host-side orchestration
// ---------------------------------------------------------------------------
static inline void gemm(const float* A, const float* W, const float* bias,
                        float* C, int M, int N, int K, int ldc, int accum,
                        hipStream_t s)
{
  dim3 g(N / 128, (M + 127) / 128);
  const bool full = (M % 128) == 0;
  if (full) {
    if (accum) gemm_wmma<true, true ><<<g, 256, 0, s>>>(A, W, bias, C, M, N, K, ldc);
    else       gemm_wmma<true, false><<<g, 256, 0, s>>>(A, W, bias, C, M, N, K, ldc);
  } else {
    if (accum) gemm_wmma<false, true ><<<g, 256, 0, s>>>(A, W, bias, C, M, N, K, ldc);
    else       gemm_wmma<false, false><<<g, 256, 0, s>>>(A, W, bias, C, M, N, K, ldc);
  }
}

struct AttnP { const float *w, *b, *pw, *pb; };
struct BlkP {
  const float *ln1w, *ln1b; AttnP at;
  const float *ln2w, *ln2b, *fcw, *fcb, *pjw, *pjb;
  const float *lncw, *lncb; AttnP cr;
};

extern "C" void kernel_launch(void* const* d_in, const int* in_sizes, int n_in,
                              void* d_out, int out_size, void* d_ws, size_t ws_size,
                              hipStream_t stream)
{
  (void)in_sizes; (void)n_in; (void)out_size; (void)ws_size;
  int ip = 0;
  auto nf = [&]() { return (const float*)d_in[ip++]; };

  const float* wte      = nf();
  const float* wte_dec  = nf();
  const float* wpe_enc  = nf();
  const float* wpe_dec  = nf();
  const float* mask_emb = nf();
  const float* linear_w = nf();
  const float* linear_b = nf();

  BlkP enc[12], dec[6];
  for (int l = 0; l < 12; ++l) {
    enc[l].ln1w = nf(); enc[l].ln1b = nf();
    enc[l].at.w = nf(); enc[l].at.b = nf(); enc[l].at.pw = nf(); enc[l].at.pb = nf();
    enc[l].ln2w = nf(); enc[l].ln2b = nf();
    enc[l].fcw = nf();  enc[l].fcb = nf();  enc[l].pjw = nf();  enc[l].pjb = nf();
    enc[l].lncw = nullptr; enc[l].lncb = nullptr;
    enc[l].cr = AttnP{nullptr, nullptr, nullptr, nullptr};
  }
  for (int l = 0; l < 6; ++l) {
    dec[l].ln1w = nf(); dec[l].ln1b = nf();
    dec[l].at.w = nf(); dec[l].at.b = nf(); dec[l].at.pw = nf(); dec[l].at.pb = nf();
    dec[l].ln2w = nf(); dec[l].ln2b = nf();
    dec[l].fcw = nf();  dec[l].fcb = nf();  dec[l].pjw = nf();  dec[l].pjb = nf();
    dec[l].lncw = nf(); dec[l].lncb = nf();
    dec[l].cr.w = nf(); dec[l].cr.b = nf(); dec[l].cr.pw = nf(); dec[l].cr.pb = nf();
  }
  const float* lnfw = nf();
  const float* lnfb = nf();
  const int* idx     = (const int*)d_in[ip++];
  const int* targets = (const int*)d_in[ip++];

  // ---- workspace carve-out (~77 MB fp32) ----
  float* w = (float*)d_ws;
  float* x0    = w; w += (size_t)M_TOK * C_EMB;       // encoder stream
  float* x1    = w; w += (size_t)M_TOK * C_EMB;       // decoder stream
  float* lnb   = w; w += (size_t)M_TOK * C_EMB;       // layernorm output
  float* abuf  = w; w += (size_t)M_TOK * C_EMB;       // attention y (pre-proj)
  float* qc    = w; w += (size_t)M_TOK * C_EMB;       // cross-attn queries
  float* qkvb  = w; w += (size_t)M_TOK * 3 * C_EMB;   // fused qkv
  float* hbuf  = w; w += (size_t)M_TOK * 4 * C_EMB;   // MLP hidden
  float* acts  = w; w += (size_t)BATCH * 5 * C_EMB;   // encoder memory (B,5,C)
  float* meanb = w; w += (size_t)BATCH * C_EMB;
  float* kvb   = w; w += (size_t)BATCH * 5 * 2 * C_EMB;
  float* attb  = w; w += (size_t)BATCH * NH * T_HALF * 5;
  float* kls   = w; w += 3;
  float* ce    = w; w += 1;

  const int EB = 256;
  const int nEmb = M_TOK * C_EMB;
  embed_kernel<<<(nEmb + EB - 1) / EB, EB, 0, stream>>>(idx, wte, wpe_enc, mask_emb, x0, 0);
  embed_kernel<<<(nEmb + EB - 1) / EB, EB, 0, stream>>>(idx, wte_dec, wpe_dec, mask_emb, x1, 1);

  const int ATTG  = BATCH * NH * (T_HALF / 128);      // 192 blocks (flash attn)
  const int XWAVG = BATCH * NH * T_HALF / 8;          // 3072 blocks (cross attn)

  // ---------------- encoder ----------------
  int slot = 0;
  for (int l = 0; l < 12; ++l) {
    const BlkP& p = enc[l];
    ln_kernel<<<M_TOK, 256, 0, stream>>>(x0, p.ln1w, p.ln1b, lnb);
    gemm(lnb, p.at.w, p.at.b, qkvb, M_TOK, 3 * C_EMB, C_EMB, 3 * C_EMB, 0, stream);
    attn_self_wmma<<<ATTG, 256, 0, stream>>>(qkvb, abuf, 0);
    gemm(abuf, p.at.pw, p.at.pb, x0, M_TOK, C_EMB, C_EMB, C_EMB, 1, stream);
    ln_kernel<<<M_TOK, 256, 0, stream>>>(x0, p.ln2w, p.ln2b, lnb);
    gemm(lnb, p.fcw, p.fcb, hbuf, M_TOK, 4 * C_EMB, C_EMB, 4 * C_EMB, 0, stream);
    gelu_kernel<<<(M_TOK * 4 * C_EMB + EB - 1) / EB, EB, 0, stream>>>(hbuf, M_TOK * 4 * C_EMB);
    gemm(hbuf, p.pjw, p.pjb, x0, M_TOK, C_EMB, 4 * C_EMB, C_EMB, 1, stream);
    if (l == 2 || l == 4 || l == 6 || l == 8 || l == 10) {
      mean_kernel<<<(BATCH * C_EMB + EB - 1) / EB, EB, 0, stream>>>(x0, meanb);
      gemm(meanb, linear_w, linear_b, acts + slot * C_EMB,
           BATCH, C_EMB, C_EMB, 5 * C_EMB, 0, stream);
      ++slot;
    }
  }

  // ---------------- decoder ----------------
  int kli = 0;
  for (int l = 0; l < 6; ++l) {
    const BlkP& p = dec[l];
    ln_kernel<<<M_TOK, 256, 0, stream>>>(x1, p.ln1w, p.ln1b, lnb);
    gemm(lnb, p.at.w, p.at.b, qkvb, M_TOK, 3 * C_EMB, C_EMB, 3 * C_EMB, 0, stream);
    attn_self_wmma<<<ATTG, 256, 0, stream>>>(qkvb, abuf, 1);
    gemm(abuf, p.at.pw, p.at.pb, x1, M_TOK, C_EMB, C_EMB, C_EMB, 1, stream);

    if (l == 1 || l == 3 || l == 5) {
      ln_kernel<<<M_TOK, 256, 0, stream>>>(x1, p.lncw, p.lncb, lnb);
      gemm(lnb, p.cr.w, p.cr.b, qc, M_TOK, C_EMB, C_EMB, C_EMB, 0, stream);
      gemm(acts, p.cr.w + (size_t)C_EMB * C_EMB, p.cr.b + C_EMB, kvb,
           BATCH * 5, 2 * C_EMB, C_EMB, 2 * C_EMB, 0, stream);
      attn_cross<<<XWAVG, 256, 0, stream>>>(qc, kvb, abuf, attb);
      gemm(abuf, p.cr.pw, p.cr.pb, x1, M_TOK, C_EMB, C_EMB, C_EMB, 1, stream);
      kl_kernel<<<1, 256, 0, stream>>>(attb, kls + kli);
      ++kli;
    }

    ln_kernel<<<M_TOK, 256, 0, stream>>>(x1, p.ln2w, p.ln2b, lnb);
    gemm(lnb, p.fcw, p.fcb, hbuf, M_TOK, 4 * C_EMB, C_EMB, 4 * C_EMB, 0, stream);
    gelu_kernel<<<(M_TOK * 4 * C_EMB + EB - 1) / EB, EB, 0, stream>>>(hbuf, M_TOK * 4 * C_EMB);
    gemm(hbuf, p.pjw, p.pjb, x1, M_TOK, C_EMB, 4 * C_EMB, C_EMB, 1, stream);
  }

  // ---------------- head: logits + loss ----------------
  float* logits = (float*)d_out;
  ln_kernel<<<M_TOK, 256, 0, stream>>>(x1, lnfw, lnfb, lnb);
  gemm(lnb, wte, nullptr, logits, M_TOK, VOCAB_N, C_EMB, VOCAB_N, 0, stream);

  zero_kernel<<<1, 32, 0, stream>>>(ce, 1);
  loss_rows<<<M_TOK, 256, 0, stream>>>(logits, targets, ce);
  finalize_kernel<<<1, 32, 0, stream>>>(ce, kls, logits + (size_t)M_TOK * VOCAB_N);
}
